// LMUModel_40853728920215
// MI455X (gfx1250) — matrix operand order repaired and checked by reading, not compiled, not run on hardware
//
#include <hip/hip_runtime.h>

typedef __attribute__((ext_vector_type(16))) _Float16 v16h;
typedef __attribute__((ext_vector_type(8)))  _Float16 v8h;
typedef __attribute__((ext_vector_type(8)))  float    v8f;

#define LMU_HIDDEN 49
#define LMU_ORDER  4
#define LMU_SEQ    4096
#define LMU_KX     53   // K-slot carrying x_t; same index as N-column carrying fused y
#define LMU_KD     64   // padded K (state width)
#define LMU_ND     64   // padded N (output width)

#if __has_builtin(__builtin_amdgcn_tanhf)
#define ACT_TANH(x) __builtin_amdgcn_tanhf(x)
#else
#define ACT_TANH(x) tanhf(x)
#endif

__global__ __launch_bounds__(32)
void lmu_fused_wmma(const float* __restrict__ x,
                    const float* __restrict__ e_x,
                    const float* __restrict__ e_h,
                    const float* __restrict__ e_m,
                    const float* __restrict__ Wx,
                    const float* __restrict__ Wh,
                    const float* __restrict__ Wm,
                    const float* __restrict__ dense_w,
                    const float* __restrict__ dense_b,
                    float* __restrict__ out)
{
    // fused weight matrix, stored n-major:  Wl[n*64 + k] = W~[k][n]  (f16)
    __shared__ _Float16 Wl[LMU_ND * LMU_KD];
    // state tile, rows = 16 batches, cols = [h(0..48) | m(49..52) | 0pad] (x injected in regs)
    __shared__ _Float16 S[16 * LMU_KD];

    const int lane = threadIdx.x;   // 0..31 (wave32)
    const int g    = blockIdx.x;    // 16-batch group, 8 groups total

    // ---- ZOH discretization: Md = expm(M), 5x5, scaling-and-squaring in f64 -------
    double E[25];
    {
        double Ms[25];
        const double SC = 1.0 / 256.0;            // scale 2^-8, undone by 8 squarings
        for (int i = 0; i < 4; ++i) {
            double Ri = (2.0 * i + 1.0) / 4.0;    // (2q+1)/THETA, THETA=4
            for (int j = 0; j < 4; ++j) {
                double a = (i < j) ? -1.0 : (((i - j) & 1) ? 1.0 : -1.0);
                Ms[i * 5 + j] = a * Ri * SC;
            }
            Ms[i * 5 + 4] = ((i & 1) ? -1.0 : 1.0) * Ri * SC;   // B column
        }
        for (int j = 0; j < 5; ++j) Ms[20 + j] = 0.0;

        double T[25];
        for (int i = 0; i < 25; ++i) { double d = (i % 6 == 0) ? 1.0 : 0.0; E[i] = d; T[i] = d; }
        for (int it = 1; it <= 13; ++it) {        // Taylor series of expm(Ms)
            double Tn[25];
            for (int i = 0; i < 5; ++i)
                for (int j = 0; j < 5; ++j) {
                    double s = 0.0;
                    for (int k = 0; k < 5; ++k) s += T[i * 5 + k] * Ms[k * 5 + j];
                    Tn[i * 5 + j] = s / (double)it;
                }
            for (int i = 0; i < 25; ++i) { T[i] = Tn[i]; E[i] += Tn[i]; }
        }
        for (int sq = 0; sq < 8; ++sq) {          // E <- E*E, 8x
            double En[25];
            for (int i = 0; i < 5; ++i)
                for (int j = 0; j < 5; ++j) {
                    double s = 0.0;
                    for (int k = 0; k < 5; ++k) s += E[i * 5 + k] * E[k * 5 + j];
                    En[i * 5 + j] = s;
                }
            for (int i = 0; i < 25; ++i) E[i] = En[i];
        }
    }
    float Adf[16], Bdf[4];
    for (int q = 0; q < 4; ++q) {
        for (int j = 0; j < 4; ++j) Adf[q * 4 + j] = (float)E[q * 5 + j];
        Bdf[q] = (float)E[q * 5 + 4];
    }
    const float ex = e_x[0];

    // ---- build fused W~ (64x64 f16): one GEMM yields h-pre, m', AND y --------------
    // D[b][n] = sum_k S[b][k]*W~[k][n] with S = [h | m | x | 0]
    //   n<49     : h-preactivation (Wh,Wm,Wx with u folded in)
    //   n=49..52 : m'  (Ad/Bd/e_* folded)
    //   n=53     : y = dense_w . h   (evaluated one step late, on the stored state)
    for (int n = lane; n < LMU_ND; n += 32) {
        float wmbd = 0.f;                          // (Wm @ Bd)[n]
        if (n < LMU_HIDDEN)
            for (int j = 0; j < 4; ++j) wmbd += Wm[n * 4 + j] * Bdf[j];
        for (int k = 0; k < LMU_KD; ++k) {
            float v = 0.f;
            if (n < LMU_HIDDEN) {
                if (k < LMU_HIDDEN) {
                    v = Wh[n * LMU_HIDDEN + k] + wmbd * e_h[k];
                } else if (k < LMU_KX) {
                    int q = k - LMU_HIDDEN;
                    float wmad = 0.f;              // (Wm @ Ad)[n][q]
                    for (int j = 0; j < 4; ++j) wmad += Wm[n * 4 + j] * Adf[j * 4 + q];
                    v = wmad + wmbd * e_m[q];
                } else if (k == LMU_KX) {
                    v = Wx[n] + wmbd * ex;
                }
            } else if (n < LMU_HIDDEN + LMU_ORDER) {
                int q = n - LMU_HIDDEN;
                if (k < LMU_HIDDEN) {
                    v = Bdf[q] * e_h[k];
                } else if (k < LMU_KX) {
                    v = Adf[q * 4 + (k - LMU_HIDDEN)] + Bdf[q] * e_m[k - LMU_HIDDEN];
                } else if (k == LMU_KX) {
                    v = Bdf[q] * ex;
                }
            } else if (n == LMU_KX) {              // fused dense head
                if (k < LMU_HIDDEN) v = dense_w[k];
            }
            Wl[n * LMU_KD + k] = (_Float16)v;
        }
    }
    // h0 = 0, m0 = 0, padding = 0 (col 53 of S stays 0 forever; x lives in registers)
    for (int i = lane; i < 16 * LMU_KD; i += 32) S[i] = (_Float16)0.f;
    __syncthreads();

    // ---- B fragments resident in VGPRs (lane = N column, K in table order per lane) ----
    const int nl   = lane & 15;
    const int hi16 = lane >> 4;
    v16h bf[4][2];
    #pragma unroll
    for (int tile = 0; tile < 4; ++tile) {
        #pragma unroll
        for (int kc = 0; kc < 2; ++kc) {
            const _Float16* wp = &Wl[(tile * 16 + nl) * LMU_KD + kc * 32 + hi16 * 8];
            v8h q0 = *(const v8h*)wp;
            v8h q1 = *(const v8h*)(wp + 16);
            bf[tile][kc] = __builtin_shufflevector(q0, q1, 0,1,2,3,4,5,6,7,8,9,10,11,12,13,14,15);
        }
    }
    const float bias = dense_b[0];

    const float* xrow = x   + ((size_t)g * 16 + nl) * LMU_SEQ;   // this lane's batch row
    const int    srow = hi16 * 8;                                // D-layout M-row base
    float*       orow = out + ((size_t)g * 16 + srow) * LMU_SEQ; // rows srow..srow+7
    const bool   ywriter = (nl == 5);                            // N column 53 = 3*16+5

    // A-fragment (16-bit 16x32) addressing into row-major S
    const int arow = nl * LMU_KD;
    const int ab0  = hi16 ? 8 : 0;
    const int ab1  = ab0 + 16;

    auto load_a = [&](v16h& a_lo, v16h& a_hi) {
        v8h p0 = *(const v8h*)&S[arow + ab0];
        v8h p1 = *(const v8h*)&S[arow + ab1];
        v8h p2 = *(const v8h*)&S[arow + 32 + ab0];
        v8h p3 = *(const v8h*)&S[arow + 32 + ab1];
        a_lo = __builtin_shufflevector(p0, p1, 0,1,2,3,4,5,6,7,8,9,10,11,12,13,14,15);
        a_hi = __builtin_shufflevector(p2, p3, 0,1,2,3,4,5,6,7,8,9,10,11,12,13,14,15);
    };

    // software-pipelined x: one b128 per 4 steps, prefetched one group ahead
    float4 xc = *(const float4*)(xrow);

    for (int tg = 0; tg < LMU_SEQ; tg += 4) {
        const int tnext = (tg + 4 < LMU_SEQ) ? (tg + 4) : (LMU_SEQ - 4);
        float4 xn = *(const float4*)(xrow + tnext);

        #pragma unroll
        for (int s = 0; s < 4; ++s) {
            const int   t  = tg + s;
            const float xv = (s == 0) ? xc.x : (s == 1) ? xc.y : (s == 2) ? xc.z : xc.w;

            v16h a_lo, a_hi;
            load_a(a_lo, a_hi);
            // inject x_t at K=53: half 13 of a_hi on lanes 0-15 (K 48..55 block);
            // lanes 16-31 keep the loaded 0 (K=61 padding).
            a_hi[13] = hi16 ? a_hi[13] : (_Float16)xv;

            v8f z = {0.f,0.f,0.f,0.f,0.f,0.f,0.f,0.f};
            v8f acc0 = z, acc1 = z, acc2 = z, acc3 = z;
            acc0 = __builtin_amdgcn_wmma_f32_16x16x32_f16(false, a_lo, false, bf[0][0], (short)0, acc0, false, false);
            acc1 = __builtin_amdgcn_wmma_f32_16x16x32_f16(false, a_lo, false, bf[1][0], (short)0, acc1, false, false);
            acc2 = __builtin_amdgcn_wmma_f32_16x16x32_f16(false, a_lo, false, bf[2][0], (short)0, acc2, false, false);
            acc3 = __builtin_amdgcn_wmma_f32_16x16x32_f16(false, a_lo, false, bf[3][0], (short)0, acc3, false, false);
            acc0 = __builtin_amdgcn_wmma_f32_16x16x32_f16(false, a_hi, false, bf[0][1], (short)0, acc0, false, false);
            acc1 = __builtin_amdgcn_wmma_f32_16x16x32_f16(false, a_hi, false, bf[1][1], (short)0, acc1, false, false);
            acc2 = __builtin_amdgcn_wmma_f32_16x16x32_f16(false, a_hi, false, bf[2][1], (short)0, acc2, false, false);
            acc3 = __builtin_amdgcn_wmma_f32_16x16x32_f16(false, a_hi, false, bf[3][1], (short)0, acc3, false, false);

            // epilogue: tanh h-cols, raw m'-cols, write state back (f16)
            #pragma unroll
            for (int tile = 0; tile < 4; ++tile) {
                const v8f acc = (tile == 0) ? acc0 : (tile == 1) ? acc1 : (tile == 2) ? acc2 : acc3;
                const int  n   = tile * 16 + nl;
                const bool isH = (n < LMU_HIDDEN);   // tanh
                const bool isS = (n < LMU_KX);       // carried state (h or m')
                #pragma unroll
                for (int j = 0; j < 8; ++j) {
                    float v  = acc[j];
                    float hv = isH ? ACT_TANH(v) : v;
                    if (isS) S[(srow + j) * LMU_KD + n] = (_Float16)hv;
                }
            }

            // y_{t-1} materializes in column 53 (this step's GEMM ran on state h_{t-1})
            if (ywriter && (t > 0)) {
                #pragma unroll
                for (int j = 0; j < 8; ++j)
                    orow[(size_t)j * LMU_SEQ + (t - 1)] = acc3[j] + bias;
            }
        }
        xc = xn;
    }

    // final output y_{SEQ-1}: one extra pass over the tile carrying column 53
    {
        v16h a_lo, a_hi;
        load_a(a_lo, a_hi);
        v8f acc3 = {0.f,0.f,0.f,0.f,0.f,0.f,0.f,0.f};
        acc3 = __builtin_amdgcn_wmma_f32_16x16x32_f16(false, a_lo, false, bf[3][0], (short)0, acc3, false, false);
        acc3 = __builtin_amdgcn_wmma_f32_16x16x32_f16(false, a_hi, false, bf[3][1], (short)0, acc3, false, false);
        if (ywriter) {
            #pragma unroll
            for (int j = 0; j < 8; ++j)
                orow[(size_t)j * LMU_SEQ + (LMU_SEQ - 1)] = acc3[j] + bias;
        }
    }
}

extern "C" void kernel_launch(void* const* d_in, const int* in_sizes, int n_in,
                              void* d_out, int out_size, void* d_ws, size_t ws_size,
                              hipStream_t stream) {
    (void)in_sizes; (void)n_in; (void)out_size; (void)d_ws; (void)ws_size;
    const float* x       = (const float*)d_in[0];
    const float* e_x     = (const float*)d_in[1];
    const float* e_h     = (const float*)d_in[2];
    const float* e_m     = (const float*)d_in[3];
    const float* Wx      = (const float*)d_in[4];
    const float* Wh      = (const float*)d_in[5];
    const float* Wm      = (const float*)d_in[6];
    const float* dense_w = (const float*)d_in[7];
    const float* dense_b = (const float*)d_in[8];
    float* out = (float*)d_out;
    // 128 batches / 16 per wave = 8 single-wave workgroups, no inter-wave sync
    lmu_fused_wmma<<<8, 32, 0, stream>>>(x, e_x, e_h, e_m, Wx, Wh, Wm,
                                         dense_w, dense_b, out);
}
